// AttentionalPropagation_44238163149252
// MI455X (gfx1250) — compile-verified, matrix operand order checked by reading, not run on hardware
//
#include <hip/hip_runtime.h>

// ---------------------------------------------------------------------------
// AttentionalPropagation for MI455X (gfx1250, wave32, WMMA bf16 16x16x32)
//   D=256 channels, H=4 heads, head-dim 64, N=M=4096
// ---------------------------------------------------------------------------

typedef __attribute__((ext_vector_type(16))) __bf16 v16bf;
typedef __attribute__((ext_vector_type(8)))  float  v8f;

#define D_DIM   256
#define H_HEADS 4
#define HD      64
#define N_Q     4096
#define M_S     4096

// ---------------------------------------------------------------------------
// helpers
// ---------------------------------------------------------------------------
union FragBF { uint4 q[2]; v16bf v; };

// Load one 16x32 bf16 WMMA operand fragment (A layout == B layout mirrored):
// lane l covers row/col (l&15); k bytes: kb..kb+7 and kb+16..kb+23, kb = (l<16)?0:8.
// Matrix is row-major along k with leading dim `ld` (multiple of 8 -> 16B aligned).
__device__ __forceinline__ v16bf load_frag(const __bf16* __restrict__ mat, int ld,
                                           int tile_rc, int k0, int lane) {
    int lr = lane & 15;
    int kb = (lane & 16) >> 1;            // 0 or 8
    FragBF f;
    const uint4* p = (const uint4*)(mat + (size_t)(tile_rc + lr) * ld + k0 + kb);
    f.q[0] = p[0];                        // k = kb .. kb+7
    f.q[1] = p[2];                        // k = kb+16 .. kb+23
    return f.v;
}

__device__ __forceinline__ v8f wmma_bf16(v16bf a, v16bf b, v8f c) {
    return __builtin_amdgcn_wmma_f32_16x16x32_bf16(false, a, false, b,
                                                   (short)0, c, false, false);
}

// monotonic float<->uint encoding so unsigned atomicMin == float min
__device__ __forceinline__ unsigned f2key(float f) {
    unsigned u = __float_as_uint(f);
    return u ^ ((unsigned)((int)u >> 31) | 0x80000000u);
}
__device__ __forceinline__ float key2f(unsigned k) {
    unsigned u = (k & 0x80000000u) ? (k ^ 0x80000000u) : ~k;
    return __uint_as_float(u);
}

// ---------------------------------------------------------------------------
// small prep kernels
// ---------------------------------------------------------------------------
__global__ void init_min_kernel(unsigned* minbuf) { *minbuf = 0xFFFFFFFFu; }

__global__ void convert_cvt(const float* __restrict__ src, __bf16* __restrict__ dst, int n) {
    int i = blockIdx.x * blockDim.x + threadIdx.x;
    if (i < n) dst[i] = (__bf16)src[i];
}

// src: [C][N] row-major fp32  ->  dst: [N][ldo] bf16 (transposed)
__global__ void transpose_cvt(const float* __restrict__ src, __bf16* __restrict__ dst,
                              int C, int N, int ldo) {
    int idx = blockIdx.x * blockDim.x + threadIdx.x;
    if (idx >= C * N) return;
    int c = idx / N, n = idx - c * N;
    dst[(size_t)n * ldo + c] = (__bf16)src[idx];
}

// ---------------------------------------------------------------------------
// generic NT GEMM:  C[row][col] = sum_k A[row][k] * B[col][k]  (+bias, relu)
// block = 256 thr = 8 waves; block tile 64 rows x 64 cols; wave -> 2 16x16 tiles
// epilogue store modes:
//   0: fp32 out[row*ldo + col]
//   1: bf16 out[col*ldo + row]                       (transposed)
//   2: bf16 out[(row&3)*hparam + col*64 + (row>>2)]  (Qt/Kt head-split, transposed)
//   3: bf16 out[(row&3)*hparam + (row>>2)*ldo + col] (Vh head-split, row-major)
//   4: bf16 out[col*ldo + row*4 + hparam]            (attnT scatter, hparam = head)
// ---------------------------------------------------------------------------
__global__ __launch_bounds__(256) void gemm_nt(
    const __bf16* __restrict__ A, int lda,
    const __bf16* __restrict__ B, int ldb,
    const float*  __restrict__ bias,
    int K, int mode, int relu,
    void* __restrict__ out, int ldo, int hparam)
{
    int lane = threadIdx.x & 31;
    int wave = threadIdx.x >> 5;
    int row0 = blockIdx.x * 64 + (wave >> 1) * 16;
    int col0 = blockIdx.y * 64 + (wave & 1) * 32;

    v8f acc0 = {}, acc1 = {};
    for (int k0 = 0; k0 < K; k0 += 32) {
        v16bf a  = load_frag(A, lda, row0, k0, lane);
        v16bf b0 = load_frag(B, ldb, col0,      k0, lane);
        v16bf b1 = load_frag(B, ldb, col0 + 16, k0, lane);
        acc0 = wmma_bf16(a, b0, acc0);
        acc1 = wmma_bf16(a, b1, acc1);
    }

    int colA  = col0 + (lane & 15);
    int colB  = colA + 16;
    int rbase = row0 + ((lane & 16) >> 1);   // +8 for lanes 16..31

    for (int r = 0; r < 8; ++r) {
        int row = rbase + r;
        float v0 = acc0[r], v1 = acc1[r];
        if (bias) { float bb = bias[row]; v0 += bb; v1 += bb; }
        if (relu) { v0 = fmaxf(v0, 0.f); v1 = fmaxf(v1, 0.f); }
        if (mode == 0) {
            float* o = (float*)out;
            o[(size_t)row * ldo + colA] = v0;
            o[(size_t)row * ldo + colB] = v1;
        } else if (mode == 1) {
            __bf16* o = (__bf16*)out;
            o[(size_t)colA * ldo + row] = (__bf16)v0;
            o[(size_t)colB * ldo + row] = (__bf16)v1;
        } else if (mode == 2) {
            __bf16* o = (__bf16*)out;
            int h = row & 3, d = row >> 2;
            o[(size_t)h * hparam + (size_t)colA * HD + d] = (__bf16)v0;
            o[(size_t)h * hparam + (size_t)colB * HD + d] = (__bf16)v1;
        } else if (mode == 3) {
            __bf16* o = (__bf16*)out;
            int h = row & 3, d = row >> 2;
            o[(size_t)h * hparam + (size_t)d * ldo + colA] = (__bf16)v0;
            o[(size_t)h * hparam + (size_t)d * ldo + colB] = (__bf16)v1;
        } else { // mode 4
            __bf16* o = (__bf16*)out;
            o[(size_t)colA * ldo + row * 4 + hparam] = (__bf16)v0;
            o[(size_t)colB * ldo + row * 4 + hparam] = (__bf16)v1;
        }
    }
}

// ---------------------------------------------------------------------------
// scores:  S[h][n][m] = (Qt_h[n][:] . Kt_h[m][:]) / 8 ; fused global-min reduce
// grid (N/64, M/64, H), block 256
// ---------------------------------------------------------------------------
__global__ __launch_bounds__(256) void scores_kernel(
    const __bf16* __restrict__ Qt, const __bf16* __restrict__ Kt,
    float* __restrict__ S, unsigned* __restrict__ minbuf)
{
    __shared__ float red[256];
    int lane = threadIdx.x & 31;
    int wave = threadIdx.x >> 5;
    int h = blockIdx.z;
    const __bf16* Qh = Qt + (size_t)h * (N_Q * HD);
    const __bf16* Kh = Kt + (size_t)h * (M_S * HD);
    float*        Sh = S  + (size_t)h * N_Q * M_S;

    int n0 = blockIdx.x * 64 + (wave >> 1) * 16;
    int m0 = blockIdx.y * 64 + (wave & 1) * 32;

    v8f acc0 = {}, acc1 = {};
    for (int k0 = 0; k0 < HD; k0 += 32) {
        v16bf a  = load_frag(Qh, HD, n0, k0, lane);
        v16bf b0 = load_frag(Kh, HD, m0,      k0, lane);
        v16bf b1 = load_frag(Kh, HD, m0 + 16, k0, lane);
        acc0 = wmma_bf16(a, b0, acc0);
        acc1 = wmma_bf16(a, b1, acc1);
    }

    int mA = m0 + (lane & 15), mB = mA + 16;
    int nb = n0 + ((lane & 16) >> 1);
    float lmin = 3.4e38f;
    for (int r = 0; r < 8; ++r) {
        float v0 = acc0[r] * 0.125f;
        float v1 = acc1[r] * 0.125f;
        Sh[(size_t)(nb + r) * M_S + mA] = v0;
        Sh[(size_t)(nb + r) * M_S + mB] = v1;
        lmin = fminf(lmin, fminf(v0, v1));
    }

    red[threadIdx.x] = lmin;
    __syncthreads();
    for (int s = 128; s > 0; s >>= 1) {
        if (threadIdx.x < s) red[threadIdx.x] = fminf(red[threadIdx.x], red[threadIdx.x + s]);
        __syncthreads();
    }
    if (threadIdx.x == 0) atomicMin(minbuf, f2key(red[0]));
}

// ---------------------------------------------------------------------------
// softmax + weights. One block per query row n; loops over 4 heads.
// Applies smin*(~mask), softmax over m, writes prob as bf16 IN-PLACE over the
// fp32 score row (block owns its 4 rows -> safe), and weights = mean_h(scores).
// ---------------------------------------------------------------------------
__global__ __launch_bounds__(256) void softmax_weights_kernel(
    float* __restrict__ S, const unsigned char* __restrict__ mask,
    const unsigned* __restrict__ minbuf, float* __restrict__ wout)
{
    __shared__ float red[256];
    int n = blockIdx.x, tid = threadIdx.x;
    float smin = key2f(*minbuf);

    float wacc[16];
    for (int i = 0; i < 16; ++i) wacc[i] = 0.f;

    for (int h = 0; h < H_HEADS; ++h) {
        float* base = S + ((size_t)h * N_Q + n) * M_S;
        float sv[16];
        float lmax = -3.4e38f;
        for (int i = 0; i < 16; ++i) {
            int m = i * 256 + tid;
            float s = base[m];
            if (!mask[(size_t)n * M_S + m]) s += smin;   // additive global-min mask
            sv[i] = s;
            lmax = fmaxf(lmax, s);
        }
        red[tid] = lmax; __syncthreads();
        for (int s2 = 128; s2 > 0; s2 >>= 1) {
            if (tid < s2) red[tid] = fmaxf(red[tid], red[tid + s2]);
            __syncthreads();
        }
        float rmax = red[0]; __syncthreads();

        float ev[16], lsum = 0.f;
        for (int i = 0; i < 16; ++i) { ev[i] = __expf(sv[i] - rmax); lsum += ev[i]; }
        red[tid] = lsum; __syncthreads();
        for (int s2 = 128; s2 > 0; s2 >>= 1) {
            if (tid < s2) red[tid] += red[tid + s2];
            __syncthreads();
        }
        float inv = 1.f / red[0]; __syncthreads();

        __bf16* P = (__bf16*)base;                       // in-place bf16 probs
        for (int i = 0; i < 16; ++i) {
            P[i * 256 + tid] = (__bf16)(ev[i] * inv);
            wacc[i] += sv[i];
        }
        __syncthreads();
    }

    for (int i = 0; i < 16; ++i)
        wout[(size_t)n * M_S + i * 256 + tid] = wacc[i] * 0.25f;
}

// ---------------------------------------------------------------------------
// host-side launch
// ---------------------------------------------------------------------------
extern "C" void kernel_launch(void* const* d_in, const int* in_sizes, int n_in,
                              void* d_out, int out_size, void* d_ws, size_t ws_size,
                              hipStream_t stream) {
    const float* x    = (const float*)d_in[0];   // [256][4096]
    const float* srcp = (const float*)d_in[1];   // [256][4096]
    // d_in[2] = dist : unused by the reference
    const unsigned char* mask = (const unsigned char*)d_in[3]; // [4096][4096] bool
    const float* Wq = (const float*)d_in[4];  const float* bq = (const float*)d_in[5];
    const float* Wk = (const float*)d_in[6];  const float* bk = (const float*)d_in[7];
    const float* Wv = (const float*)d_in[8];  const float* bv = (const float*)d_in[9];
    const float* Wm = (const float*)d_in[10]; const float* bm = (const float*)d_in[11];
    const float* W1 = (const float*)d_in[12]; const float* b1 = (const float*)d_in[13];
    const float* W2 = (const float*)d_in[14]; const float* b2 = (const float*)d_in[15];

    // ---- workspace carve-up (256B aligned slices) ----
    char* w = (char*)d_ws;
    size_t off = 0;
    auto alloc = [&](size_t bytes) -> char* {
        char* p = w + off;
        off += (bytes + 255) & ~(size_t)255;
        return p;
    };
    unsigned* minbuf = (unsigned*)alloc(256);
    __bf16* WqB  = (__bf16*)alloc((size_t)256 * 256 * 2);
    __bf16* WkB  = (__bf16*)alloc((size_t)256 * 256 * 2);
    __bf16* WvB  = (__bf16*)alloc((size_t)256 * 256 * 2);
    __bf16* WmB  = (__bf16*)alloc((size_t)256 * 256 * 2);
    __bf16* W1B  = (__bf16*)alloc((size_t)512 * 512 * 2);
    __bf16* W2B  = (__bf16*)alloc((size_t)256 * 512 * 2);
    __bf16* catT = (__bf16*)alloc((size_t)N_Q * 512 * 2);   // [n][512]: x | message
    __bf16* St   = (__bf16*)alloc((size_t)M_S * 256 * 2);   // source^T
    __bf16* Qt   = (__bf16*)alloc((size_t)H_HEADS * N_Q * HD * 2);
    __bf16* Kt   = (__bf16*)alloc((size_t)H_HEADS * M_S * HD * 2);
    __bf16* Vh   = (__bf16*)alloc((size_t)H_HEADS * HD * M_S * 2);
    __bf16* attnT= (__bf16*)alloc((size_t)N_Q * 256 * 2);
    __bf16* h1T  = (__bf16*)alloc((size_t)N_Q * 512 * 2);
    float*  Sbuf = (float*) alloc((size_t)H_HEADS * N_Q * M_S * 4); // scores / probs

    float* y_out = (float*)d_out;                 // [256][4096]
    float* w_out = (float*)d_out + (size_t)D_DIM * N_Q;  // [4096][4096]

    const size_t HSTRIDE = (size_t)N_Q * HD;      // 262144 elems per head

    // 0) init global-min accumulator
    init_min_kernel<<<1, 1, 0, stream>>>(minbuf);

    // 1) weight conversions + activation transposes
    convert_cvt<<<256, 256, 0, stream>>>(Wq, WqB, 256 * 256);
    convert_cvt<<<256, 256, 0, stream>>>(Wk, WkB, 256 * 256);
    convert_cvt<<<256, 256, 0, stream>>>(Wv, WvB, 256 * 256);
    convert_cvt<<<256, 256, 0, stream>>>(Wm, WmB, 256 * 256);
    convert_cvt<<<1024, 256, 0, stream>>>(W1, W1B, 512 * 512);
    convert_cvt<<<512, 256, 0, stream>>>(W2, W2B, 256 * 512);
    transpose_cvt<<<4096, 256, 0, stream>>>(x,    catT, 256, N_Q, 512);
    transpose_cvt<<<4096, 256, 0, stream>>>(srcp, St,   256, M_S, 256);

    // 2) Q/K/V projections (WMMA), head-split scatter epilogues
    gemm_nt<<<dim3(4, 64), 256, 0, stream>>>(WqB, 256, catT, 512, bq, 256, 2, 0, Qt, HD,  (int)HSTRIDE);
    gemm_nt<<<dim3(4, 64), 256, 0, stream>>>(WkB, 256, St,   256, bk, 256, 2, 0, Kt, HD,  (int)HSTRIDE);
    gemm_nt<<<dim3(4, 64), 256, 0, stream>>>(WvB, 256, St,   256, bv, 256, 3, 0, Vh, M_S, (int)HSTRIDE);

    // 3) scores (WMMA) + fused global min
    scores_kernel<<<dim3(64, 64, H_HEADS), 256, 0, stream>>>(Qt, Kt, Sbuf, minbuf);

    // 4) mask + softmax (probs bf16 in-place) + weights output
    softmax_weights_kernel<<<N_Q, 256, 0, stream>>>(Sbuf, mask, minbuf, w_out);

    // 5) attention aggregation per head (WMMA, K = 4096); probs read as bf16
    //    row stride of prob rows = 4096 floats = 8192 bf16 elems
    for (int h = 0; h < H_HEADS; ++h) {
        const __bf16* Ph = (const __bf16*)(Sbuf + (size_t)h * N_Q * M_S);
        gemm_nt<<<dim3(1, 64), 256, 0, stream>>>(Vh + (size_t)h * HSTRIDE, M_S,
                                                 Ph, 2 * M_S, nullptr,
                                                 M_S, 4, 0, attnT, 256, h);
    }

    // 6) merge heads: message = Wm @ attn + bm  -> catT columns 256..511
    gemm_nt<<<dim3(4, 64), 256, 0, stream>>>(WmB, 256, attnT, 256, bm, 256, 1, 0,
                                             (void*)(catT + 256), 512, 0);

    // 7) MLP: h = relu(W1 @ [x;message] + b1) ; y = W2 @ h + b2
    gemm_nt<<<dim3(8, 64), 256, 0, stream>>>(W1B, 512, catT, 512, b1, 512, 1, 1, h1T, 512, 0);
    gemm_nt<<<dim3(4, 64), 256, 0, stream>>>(W2B, 512, h1T,  512, b2, 512, 0, 0, y_out, N_Q, 0);
}
